// MotionCompFVC_19988777795682
// MI455X (gfx1250) — matrix operand order repaired and checked by reading, not compiled, not run on hardware
//
#include <hip/hip_runtime.h>

typedef __attribute__((ext_vector_type(16))) _Float16 v16h;
typedef __attribute__((ext_vector_type(8)))  _Float16 v8h;
typedef __attribute__((ext_vector_type(8)))  float    v8f;
typedef __attribute__((ext_vector_type(4)))  unsigned int v4u;
typedef __attribute__((ext_vector_type(8)))  int      v8i;
typedef __attribute__((ext_vector_type(4)))  int      v4i;

#define HH 256
#define WW 256

// ---------------------------------------------------------------------------
// fp32 -> f16 weight pre-conversion (run once per weight tensor)
// ---------------------------------------------------------------------------
__global__ __launch_bounds__(256)
void f32_to_f16_kernel(const float* __restrict__ src, _Float16* __restrict__ dst, int n)
{
    int i = blockIdx.x * 256 + threadIdx.x;
    if (i < n) dst[i] = (_Float16)src[i];
}

// ---------------------------------------------------------------------------
// Implicit-GEMM 3x3 SAME conv via v_wmma_f32_16x16x32_f16.
//   M = out channels (16/wave), N = 32 consecutive W pixels (2 wmma/wave),
//   K = CIN*9.  4 waves / 128 threads per block.
//   Weight tiles (64 x 32 f16, row stride KTOT) are DMA'd into LDS by the
//   Tensor Data Mover with pad_enable giving an 80-byte LDS row pitch
//   (bank-conflict-free fragment reads).  Wave 0 issues the TDM descriptor
//   and waits on TENSORcnt; __syncthreads publishes the tile.
// ---------------------------------------------------------------------------
template<int CIN, int COUT, bool RELU, bool DUAL, bool FINAL>
__global__ __launch_bounds__(128)
void conv3x3_wmma(const float* __restrict__ in0, const float* __restrict__ in1,
                  const _Float16* __restrict__ wgt,   // [COUT][KTOT] f16
                  const float* __restrict__ bias,
                  const float* __restrict__ addend, float* __restrict__ out)
{
    constexpr int KTOT = CIN * 9;
    constexpr int CSRC = DUAL ? (CIN / 2) : CIN;

    // 40-half row pitch (80 B): TDM pads weight rows; act tile padded manually
    __shared__ _Float16 sAct[32][40];   // [n][k] im2col B-tile
    __shared__ _Float16 sWgt[64][40];   // [m][k] A-tiles (TDM destination)

    const int tid   = threadIdx.x;
    const int lane  = tid & 31;
    const int wave  = tid >> 5;
    const int khalf = lane >> 4;
    const int nlan  = lane & 15;

    // spatial tile: 32 consecutive w positions of one (b, h) row
    const int t  = blockIdx.x;
    const int w0 = (t & 7) * 32;
    const int h  = (t >> 3) & 255;
    const int b  = t >> 11;

    const int  rowBase = blockIdx.y * 64;
    const int  mBase   = rowBase + wave * 16;     // uniform per wave
    const bool active  = (mBase < COUT);
    const bool issuer  = (tid < 32);              // wave 0 drives the TDM

    v8f acc0 = {}, acc1 = {};
    if (active) {
#pragma unroll
        for (int v = 0; v < 8; ++v) {
            float bv = bias[mBase + khalf * 8 + v];
            acc0[v] = bv; acc1[v] = bv;
        }
    }

    const unsigned ldsOff = (unsigned)(uintptr_t)(&sWgt[0][0]);
    int remM = COUT - rowBase; if (remM > 64) remM = 64;

    for (int k0 = 0; k0 < KTOT; k0 += 32) {
        // ---- cooperative im2col fill: 32 n x 32 k (8 elems/thread).
        //      idx = e*128 + tid  =>  consecutive lanes hit consecutive x
#pragma unroll
        for (int e = 0; e < 8; ++e) {
            int idx = e * 128 + tid;        // 0..1023
            int kk  = idx >> 5;
            int n   = idx & 31;
            int k   = k0 + kk;
            int c   = k / 9;
            int r   = k - c * 9;            // ky*3 + kx
            int y   = h + r / 3 - 1;
            int x   = w0 + n + (r % 3) - 1;
            float v = 0.f;
            if (y >= 0 && y < HH && x >= 0 && x < WW) {
                const float* src = in0;
                int cc = c;
                if (DUAL && c >= CSRC) { src = in1; cc = c - CSRC; }
                v = src[(((size_t)b * CSRC + cc) * HH + y) * WW + x];
            }
            sAct[n][kk] = (_Float16)v;
        }
        // ---- TDM: async-load 64x32 f16 weight tile into LDS (wave 0 only)
        if (issuer) {
            unsigned long long ga =
                (unsigned long long)(uintptr_t)(wgt + (size_t)rowBase * KTOT + k0);
            v4u g0;
            g0[0] = 1u;                               // count=1 (valid user D#)
            g0[1] = ldsOff;                           // lds_addr
            g0[2] = (unsigned)ga;                     // global_addr[31:0]
            g0[3] = (unsigned)(ga >> 32) | 0x80000000u; // addr[56:32] | type=2
            v8i g1;
            g1[0] = (1 << 16)        // data_size = 1 (2 bytes)
                  | (1 << 20)        // pad_enable
                  | (3 << 22)        // pad_interval: 16 DWORDs (one 64B row)
                  | (3 << 25);       // pad_amount: 4 DWORDs (16B) -> 80B pitch
            int td0 = KTOT - k0;                      // remaining columns
            g1[1] = (td0 & 0xFFFF) << 16;             // tensor_dim0 lo16
            g1[2] = ((td0 >> 16) & 0xFFFF) | (remM << 16); // dim0 hi | dim1 lo
            g1[3] = (32 << 16);                       // dim1 hi=0 | tile_dim0=32
            g1[4] = 64;                               // tile_dim1=64, tile_dim2=0
            g1[5] = KTOT;                             // tensor_dim0_stride lo32
            g1[6] = 0;
            g1[7] = 0;
            v4i z4 = {};
            v8i z8 = {};
            __builtin_amdgcn_tensor_load_to_lds(g0, g1, z4, z4, z8, 0);
            __builtin_amdgcn_s_wait_tensorcnt(0);
        }
        __syncthreads();

        if (active) {
            // A fragment: row M = wave*16+nlan; K = kh*16..kh*16+15
            const v8h* ar = (const v8h*)&sWgt[wave * 16 + nlan][0]; // 5 chunks/row
            v8h a0 = ar[khalf * 2];
            v8h a1 = ar[khalf * 2 + 1];
            v16h A = __builtin_shufflevector(a0, a1,
                     0,1,2,3,4,5,6,7,8,9,10,11,12,13,14,15);
            // B fragments: columns nlan and 16+nlan; K = kh*16..kh*16+15
            const v8h* b0r = (const v8h*)&sAct[nlan][0];
            const v8h* b1r = (const v8h*)&sAct[16 + nlan][0];
            v16h B0 = __builtin_shufflevector(b0r[khalf*2], b0r[khalf*2+1],
                      0,1,2,3,4,5,6,7,8,9,10,11,12,13,14,15);
            v16h B1 = __builtin_shufflevector(b1r[khalf*2], b1r[khalf*2+1],
                      0,1,2,3,4,5,6,7,8,9,10,11,12,13,14,15);
            acc0 = __builtin_amdgcn_wmma_f32_16x16x32_f16(
                       false, A, false, B0, (short)0, acc0, false, false);
            acc1 = __builtin_amdgcn_wmma_f32_16x16x32_f16(
                       false, A, false, B1, (short)0, acc1, false, false);
        }
        __syncthreads();    // nobody reads sWgt/sAct when next TDM/fill starts
    }

    if (active) {
        const size_t HW = (size_t)HH * WW;
        size_t sp = (size_t)h * WW + w0 + nlan;
#pragma unroll
        for (int v = 0; v < 8; ++v) {
            int   m  = mBase + khalf * 8 + v;
            size_t o = ((size_t)b * COUT + m) * HW + sp;
            float r0 = acc0[v], r1 = acc1[v];
            if (RELU) { r0 = fmaxf(r0, 0.f); r1 = fmaxf(r1, 0.f); }
            if constexpr (FINAL) { r0 += addend[o]; r1 += addend[o + 16]; }
            out[o]      = r0;
            out[o + 16] = r1;
        }
    }
}

// ---------------------------------------------------------------------------
// Grouped deformable conv (G=8, Cg=8, 3x3, bilinear). One thread per output
// pixel of one (b, g); group weights (8x8x9) staged in LDS.
// ---------------------------------------------------------------------------
__device__ __forceinline__ int iclamp(int v, int lo, int hi) {
    return v < lo ? lo : (v > hi ? hi : v);
}

__global__ __launch_bounds__(256)
void deform_conv_kernel(const float* __restrict__ ref,
                        const float* __restrict__ offs,
                        const float* __restrict__ wdef,
                        const float* __restrict__ bdef,
                        float* __restrict__ out)
{
    __shared__ float lw[576];   // [o][i][k2] for this group
    __shared__ float lb[8];

    const int blk   = blockIdx.x;        // (b, g, ptile)
    const int ptile = blk & 255;
    const int g     = (blk >> 8) & 7;
    const int b     = blk >> 11;
    const int tid   = threadIdx.x;

    for (int i = tid; i < 576; i += 256) lw[i] = wdef[g * 576 + i];
    if (tid < 8) lb[tid] = bdef[g * 8 + tid];
    __syncthreads();

    const int p = ptile * 256 + tid;
    const int h = p >> 8;
    const int w = p & 255;

    float acc[8];
#pragma unroll
    for (int o = 0; o < 8; ++o) acc[o] = lb[o];

    const size_t HW   = (size_t)HH * WW;
    const float* refg = ref  + ((size_t)b * 64 + g * 8) * HW;
    const float* offb = offs + (size_t)b * 144 * HW + (size_t)h * WW + w;

    for (int k2 = 0; k2 < 9; ++k2) {
        float dy = offb[(size_t)((g * 9 + k2) * 2 + 0) * HW];
        float dx = offb[(size_t)((g * 9 + k2) * 2 + 1) * HW];
        float py = dy + (float)(h + k2 / 3 - 1);
        float px = dx + (float)(w + k2 % 3 - 1);
        float y0f = floorf(py), x0f = floorf(px);
        int   y0  = (int)y0f,   x0  = (int)x0f;
        float ly  = py - y0f,   lx  = px - x0f;
        int   y1  = y0 + 1,     x1  = x0 + 1;
        bool vy0 = (y0 >= 0) & (y0 < HH), vy1 = (y1 >= 0) & (y1 < HH);
        bool vx0 = (x0 >= 0) & (x0 < WW), vx1 = (x1 >= 0) & (x1 < WW);
        float m00 = (vy0 & vx0) ? (1.f - ly) * (1.f - lx) : 0.f;
        float m01 = (vy0 & vx1) ? (1.f - ly) * lx         : 0.f;
        float m10 = (vy1 & vx0) ? ly * (1.f - lx)         : 0.f;
        float m11 = (vy1 & vx1) ? ly * lx                 : 0.f;
        size_t i00 = (size_t)iclamp(y0,0,HH-1) * WW + iclamp(x0,0,WW-1);
        size_t i01 = (size_t)iclamp(y0,0,HH-1) * WW + iclamp(x1,0,WW-1);
        size_t i10 = (size_t)iclamp(y1,0,HH-1) * WW + iclamp(x0,0,WW-1);
        size_t i11 = (size_t)iclamp(y1,0,HH-1) * WW + iclamp(x1,0,WW-1);
#pragma unroll
        for (int i = 0; i < 8; ++i) {
            const float* rc = refg + i * HW;
            float v = m00 * rc[i00] + m01 * rc[i01]
                    + m10 * rc[i10] + m11 * rc[i11];
#pragma unroll
            for (int o = 0; o < 8; ++o)
                acc[o] = fmaf(v, lw[(o * 8 + i) * 9 + k2], acc[o]);
        }
    }

    float* og = out + ((size_t)b * 64 + g * 8) * HW + (size_t)h * WW + w;
#pragma unroll
    for (int o = 0; o < 8; ++o) og[o * HW] = acc[o];
}

// ---------------------------------------------------------------------------
extern "C" void kernel_launch(void* const* d_in, const int* in_sizes, int n_in,
                              void* d_out, int out_size, void* d_ws, size_t ws_size,
                              hipStream_t stream)
{
    const float* offset = (const float*)d_in[0];
    const float* ref    = (const float*)d_in[1];
    const float* w_off  = (const float*)d_in[2];
    const float* b_off  = (const float*)d_in[3];
    const float* w_def  = (const float*)d_in[4];
    const float* b_def  = (const float*)d_in[5];
    const float* w_r1   = (const float*)d_in[6];
    const float* b_r1   = (const float*)d_in[7];
    const float* w_r2   = (const float*)d_in[8];
    const float* b_r2   = (const float*)d_in[9];
    float* out = (float*)d_out;

    const size_t HW = (size_t)HH * WW;
    // workspace: region A (144ch; deform_offsets, reused as x1), aligned_ref,
    // then f16 weight copies for the TDM path.
    float*    regA    = (float*)d_ws;
    float*    aligned = regA + (size_t)2 * 144 * HW;
    _Float16* whOff   = (_Float16*)(aligned + (size_t)2 * 64 * HW);
    _Float16* whR1    = whOff + 144 * 576;
    _Float16* whR2    = whR1  + 64 * 1152;
    float*    defOffs = regA;
    float*    x1      = regA;

    // 0) one-shot fp32 -> f16 weight conversion
    f32_to_f16_kernel<<<(144*576 + 255)/256, 256, 0, stream>>>(w_off, whOff, 144*576);
    f32_to_f16_kernel<<<(64*1152 + 255)/256, 256, 0, stream>>>(w_r1, whR1, 64*1152);
    f32_to_f16_kernel<<<(64*576  + 255)/256, 256, 0, stream>>>(w_r2, whR2, 64*576);

    const dim3 blk(128);
    const int  spTiles = 2 * HH * (WW / 32);       // 4096

    // 1) offset conv: 64 -> 144
    conv3x3_wmma<64, 144, false, false, false>
        <<<dim3(spTiles, 3), blk, 0, stream>>>(offset, nullptr, whOff, b_off,
                                               nullptr, defOffs);
    // 2) deformable conv -> aligned_ref
    deform_conv_kernel<<<dim3(4096), dim3(256), 0, stream>>>(
        ref, defOffs, w_def, b_def, aligned);
    // 3) concat([ref, aligned]) -> conv 128 -> 64, ReLU
    conv3x3_wmma<128, 64, true, true, false>
        <<<dim3(spTiles, 1), blk, 0, stream>>>(ref, aligned, whR1, b_r1,
                                               nullptr, x1);
    // 4) conv 64 -> 64, ReLU, + aligned residual
    conv3x3_wmma<64, 64, true, false, true>
        <<<dim3(spTiles, 1), blk, 0, stream>>>(x1, nullptr, whR2, b_r2,
                                               aligned, out);
}
